// MixedPrecisionLinear_42623255446083
// MI455X (gfx1250) — compile-verified, matrix operand order checked by reading, not run on hardware
//
#include <hip/hip_runtime.h>
#include <hip/hip_bf16.h>
#include <stdint.h>

typedef __attribute__((ext_vector_type(16))) __bf16 v16bf;
typedef __attribute__((ext_vector_type(8)))  float  v8f;

#define IN_DIM   4096
#define OUT_DIM  11008
#define NROWS    4096
#define GROUP_K  128                 // IN_DIM / 32 groups
#define ZCOLS    (OUT_DIM / 8)       // 1376

#define BM   128
#define BN   128
#define BK   32
#define NKT  (IN_DIM / BK)           // 128 K-tiles
#define LDSS 40                      // padded LDS row stride (bf16 elems) = 80B
#define TILE_ELEMS (BM * LDSS)       // 5120 shorts = 10KB per tile buffer

// Round-half-up f32->bf16 pair pack: one v_add each + one v_perm_b32.
// sel 0x07060302: out bytes {hi.b3, hi.b2, lo.b3, lo.b2}
__device__ __forceinline__ unsigned pack_bf16(float lo, float hi) {
  union { float f; unsigned u; } a, b;
  a.f = lo; b.f = hi;
  return __builtin_amdgcn_perm(b.u + 0x8000u, a.u + 0x8000u, 0x07060302u);
}

__global__ __launch_bounds__(256)
void gptq_bf16_wmma(const float* __restrict__ x,
                    const int*   __restrict__ qweight,
                    const int*   __restrict__ qzeros,
                    const float* __restrict__ scales,
                    const float* __restrict__ bias,
                    float*       __restrict__ out) {
  // double-buffered tiles: A = x [m][k] bf16, B = W [n][k] bf16 (transposed)
  __shared__ __align__(16) unsigned short Abuf[2 * TILE_ELEMS];
  __shared__ __align__(16) unsigned short Bbuf[2 * TILE_ELEMS];

  const int tid   = threadIdx.x;
  const int lane  = tid & 31;
  const int wave  = tid >> 5;
  const int waveM = wave & 3;              // 4 waves along M -> 32 rows each
  const int waveN = wave >> 2;             // 2 waves along N -> 64 cols each
  const int blockM = blockIdx.x * BM;
  const int blockN = blockIdx.y * BN;

  const int lanehalf = lane >> 4;          // 0 | 1
  const int khalf    = lanehalf * 8;       // per-lane K offset (16-bit A/B frag layout)
  const int mrow     = waveM * 32 + (lane & 15);
  const int nrow     = waveN * 64 + (lane & 15);

  v8f acc[2][4];
  const v8f vzero = {0.f,0.f,0.f,0.f,0.f,0.f,0.f,0.f};
  #pragma unroll
  for (int i = 0; i < 2; ++i)
    #pragma unroll
    for (int j = 0; j < 4; ++j) acc[i][j] = vzero;

  // ---- per-thread staging state (register pipeline, depth 2) ----
  float4 aReg[4];
  int    qReg[2];
  float  sReg[2], zbReg[2];                // scale and (-zero*scale)

  auto loadRegs = [&](int kt) {
    const int k0 = kt * BK;
    const int g  = k0 / GROUP_K;
    #pragma unroll
    for (int e = 0; e < 4; ++e) {
      int idx4 = tid + e * 256;            // 1024 float4 = 128 rows x 32 K
      int row  = idx4 >> 3;
      int kq   = idx4 & 7;
      aReg[e] = *(const float4*)(x + (size_t)(blockM + row) * IN_DIM + k0 + kq * 4);
    }
    #pragma unroll
    for (int e = 0; e < 2; ++e) {
      int idx = tid + e * 256;             // 512 int32 = 4 packed-K rows x 128 cols
      int rL  = idx >> 7;
      int n   = idx & 127;
      int gn  = blockN + n;
      qReg[e] = qweight[(size_t)(k0 / 8 + rL) * OUT_DIM + gn];
      int zw  = qzeros[g * ZCOLS + (gn >> 3)];
      float z = (float)((zw >> ((gn & 7) * 4)) & 0xF);
      float s = scales[(size_t)g * OUT_DIM + gn];
      sReg[e]  = s;
      zbReg[e] = -z * s;
    }
  };

  auto storeLDS = [&](int buf) {
    unsigned short* Ab = Abuf + buf * TILE_ELEMS;
    unsigned short* Bb = Bbuf + buf * TILE_ELEMS;
    #pragma unroll
    for (int e = 0; e < 4; ++e) {
      int idx4 = tid + e * 256;
      int row  = idx4 >> 3;
      int kq   = idx4 & 7;
      uint2 p;
      p.x = pack_bf16(aReg[e].x, aReg[e].y);
      p.y = pack_bf16(aReg[e].z, aReg[e].w);
      *(uint2*)(&Ab[row * LDSS + kq * 4]) = p;
    }
    #pragma unroll
    for (int e = 0; e < 2; ++e) {
      int idx = tid + e * 256;
      int rL  = idx >> 7;
      int n   = idx & 127;
      int   q  = qReg[e];
      float s  = sReg[e];
      float zb = zbReg[e];
      uint4 pk;
      {
        float v0 = fmaf((float)((q      ) & 0xF), s, zb);
        float v1 = fmaf((float)((q >>  4) & 0xF), s, zb);
        float v2 = fmaf((float)((q >>  8) & 0xF), s, zb);
        float v3 = fmaf((float)((q >> 12) & 0xF), s, zb);
        float v4 = fmaf((float)((q >> 16) & 0xF), s, zb);
        float v5 = fmaf((float)((q >> 20) & 0xF), s, zb);
        float v6 = fmaf((float)((q >> 24) & 0xF), s, zb);
        float v7 = fmaf((float)((unsigned)q >> 28), s, zb);
        pk.x = pack_bf16(v0, v1);
        pk.y = pack_bf16(v2, v3);
        pk.z = pack_bf16(v4, v5);
        pk.w = pack_bf16(v6, v7);
      }
      *(uint4*)(&Bb[n * LDSS + rL * 8]) = pk;   // local K = rL*8 .. rL*8+7, 16B store
    }
  };

  // ---- pipeline prologue ----
  loadRegs(0);
  storeLDS(0);
  loadRegs(1);

  // ---- main loop: 1 barrier / K-step; globals for kt+2 overlap WMMAs of kt ----
  for (int kt = 0; kt < NKT; ++kt) {
    __syncthreads();                       // buf[kt&1] visible; buf[(kt+1)&1] free
    if (kt + 1 < NKT) storeLDS((kt + 1) & 1);
    if (kt + 2 < NKT) loadRegs(kt + 2);

    const unsigned short* Ab = Abuf + (kt & 1) * TILE_ELEMS;
    const unsigned short* Bb = Bbuf + (kt & 1) * TILE_ELEMS;
    #pragma unroll
    for (int mt = 0; mt < 2; ++mt) {
      union { uint4 q[2]; v16bf v; } aF;
      const unsigned short* ap = &Ab[(mrow + mt * 16) * LDSS + khalf];
      aF.q[0] = *(const uint4*)(ap);       // K = khalf .. khalf+7
      aF.q[1] = *(const uint4*)(ap + 16);  // K = 16+khalf .. 16+khalf+7
      #pragma unroll
      for (int nt = 0; nt < 4; ++nt) {
        union { uint4 q[2]; v16bf v; } bF;
        const unsigned short* bp = &Bb[(nrow + nt * 16) * LDSS + khalf];
        bF.q[0] = *(const uint4*)(bp);
        bF.q[1] = *(const uint4*)(bp + 16);
        acc[mt][nt] = __builtin_amdgcn_wmma_f32_16x16x32_bf16(
            false, aF.v, false, bF.v, (short)0, acc[mt][nt], false, false);
      }
    }
  }

  // ---- epilogue: C layout VGPR v, lanes 0-15 -> M=v, lanes 16-31 -> M=v+8 ----
  #pragma unroll
  for (int nt = 0; nt < 4; ++nt) {
    int col = blockN + waveN * 64 + nt * 16 + (lane & 15);
    float bv = bias[col];
    #pragma unroll
    for (int mt = 0; mt < 2; ++mt) {
      #pragma unroll
      for (int v = 0; v < 8; ++v) {
        int row = blockM + waveM * 32 + mt * 16 + lanehalf * 8 + v;
        out[(size_t)row * OUT_DIM + col] = acc[mt][nt][v] + bv;
      }
    }
  }
}

extern "C" void kernel_launch(void* const* d_in, const int* in_sizes, int n_in,
                              void* d_out, int out_size, void* d_ws, size_t ws_size,
                              hipStream_t stream) {
  const float* x       = (const float*)d_in[0];
  const int*   qweight = (const int*)  d_in[1];
  const int*   qzeros  = (const int*)  d_in[2];
  const float* scales  = (const float*)d_in[3];
  const float* bias    = (const float*)d_in[4];
  float*       out     = (float*)d_out;
  (void)in_sizes; (void)n_in; (void)out_size; (void)d_ws; (void)ws_size;

  dim3 grid(NROWS / BM, OUT_DIM / BN);     // (32, 86)
  gptq_bf16_wmma<<<grid, 256, 0, stream>>>(x, qweight, qzeros, scales, bias, out);
}
